// Net_SSL_38740605010537
// MI455X (gfx1250) — matrix operand order repaired and checked by reading, not compiled, not run on hardware
//
#include <hip/hip_runtime.h>
#include <math.h>

#define NN 100000
#define NE 1600000
#define FIN 256
#define HID 64
#define NCLS 40

typedef __attribute__((ext_vector_type(16))) _Float16 v16h;
typedef __attribute__((ext_vector_type(8)))  float    v8f;

// Guaranteed hardware scatter-add (GLOBAL_ATOMIC_ADD_F32, no return -> STOREcnt)
__device__ __forceinline__ void gadd_f32(float* p, float v) {
    asm volatile("global_atomic_add_f32 %0, %1, off" : : "v"(p), "v"(v) : "memory");
}

// ---------------------------------------------------------------- degrees
__global__ void deg_kernel(const int* __restrict__ dst, int* __restrict__ deg, int E) {
    int i = blockIdx.x * blockDim.x + threadIdx.x;
    if (i < E) atomicAdd(&deg[dst[i]], 1);
}

__global__ void dinv_kernel(const int* __restrict__ deg, float* __restrict__ dinv, int N) {
    int i = blockIdx.x * blockDim.x + threadIdx.x;
    if (i < N) dinv[i] = rsqrtf((float)(deg[i] + 1));  // +1 self-loop
}

// ---------------------------------------------------------------- pre-pack W1 into WMMA B-fragment layout (f16)
// fragment f = (nt*8 + kc)*32 + lane holds 16 halfs:
//   lanes 0-15: K = kc*32 + 0..15 of column nt*16+(lane&15)
//   lanes16-31: K = kc*32 + 16..31
__global__ void pack_w1_kernel(const float* __restrict__ W, _Float16* __restrict__ P) {
    int t = blockIdx.x * blockDim.x + threadIdx.x;      // 0..1023
    if (t >= 4 * 8 * 32) return;
    const int lane = t & 31;
    const int kc   = (t >> 5) & 7;
    const int nt   = t >> 8;
    const int col  = nt * 16 + (lane & 15);
    const int kb   = kc * 32 + (lane >> 4) * 16;
    _Float16* o = P + (size_t)t * 16;
#pragma unroll
    for (int i = 0; i < 16; ++i) o[i] = (_Float16)W[(size_t)(kb + i) * HID + col];
}

// ---------------------------------------------------------------- pre-pack W2 (64x40 -> N padded to 48, zeros)
__global__ void pack_w2_kernel(const float* __restrict__ W, _Float16* __restrict__ P) {
    int t = blockIdx.x * blockDim.x + threadIdx.x;      // 0..191
    if (t >= 3 * 2 * 32) return;
    const int lane = t & 31;
    const int kc   = (t >> 5) & 1;
    const int nt   = t >> 6;
    const int col  = nt * 16 + (lane & 15);
    const int kb   = kc * 32 + (lane >> 4) * 16;
    _Float16* o = P + (size_t)t * 16;
#pragma unroll
    for (int i = 0; i < 16; ++i)
        o[i] = (col < NCLS) ? (_Float16)W[(size_t)(kb + i) * NCLS + col] : (_Float16)0.0f;
}

// ---------------------------------------------------------------- GEMM1: H1 = X(100000x256) * W1(256x64)
__global__ __launch_bounds__(128) void gemm1_wmma(const float* __restrict__ X,
                                                  const _Float16* __restrict__ BP,
                                                  float* __restrict__ H) {
    const int lane = threadIdx.x & 31;
    const int nt   = threadIdx.x >> 5;   // 0..3 (N tile)
    const int mt   = blockIdx.x;         // 0..6249 (M tile)
    const int l15  = lane & 15;
    const int hi   = lane >> 4;          // 0 or 1
    const int row  = mt * 16 + l15;      // A: lane = row

    v8f acc = {};
#pragma unroll
    for (int kc = 0; kc < 8; ++kc) {
        const int k0 = kc * 32;
        // A fragment: lanes0-15 K=k0..+15; lanes16-31 K=k0+16..+31
        const float* ap = X + (size_t)row * FIN + k0 + hi * 16;
        float4 f0 = *(const float4*)(ap + 0);
        float4 f1 = *(const float4*)(ap + 4);
        float4 f2 = *(const float4*)(ap + 8);
        float4 f3 = *(const float4*)(ap + 12);
        v16h a;
        a[0]=(_Float16)f0.x; a[1]=(_Float16)f0.y; a[2]=(_Float16)f0.z; a[3]=(_Float16)f0.w;
        a[4]=(_Float16)f1.x; a[5]=(_Float16)f1.y; a[6]=(_Float16)f1.z; a[7]=(_Float16)f1.w;
        a[8]=(_Float16)f2.x; a[9]=(_Float16)f2.y; a[10]=(_Float16)f2.z; a[11]=(_Float16)f2.w;
        a[12]=(_Float16)f3.x; a[13]=(_Float16)f3.y; a[14]=(_Float16)f3.z; a[15]=(_Float16)f3.w;
        // B fragment: pre-packed, 32B contiguous per lane
        const v16h b = *(const v16h*)(BP + (size_t)((nt * 8 + kc) * 32 + lane) * 16);

        acc = __builtin_amdgcn_wmma_f32_16x16x32_f16(false, a, false, b,
                                                     (short)0, acc, false, false);
    }
    // C/D: VGPR r -> row (r + 8*hi), col = lane&15
    float* hp = H + (size_t)(mt * 16 + hi * 8) * HID + nt * 16 + l15;
#pragma unroll
    for (int r = 0; r < 8; ++r) hp[(size_t)r * HID] = acc[r];
}

// ---------------------------------------------------------------- edge aggregation, 64 features (float4 chunks, 16/edge)
__global__ void agg64_kernel(const int* __restrict__ src, const int* __restrict__ dst,
                             const float* __restrict__ dinv, const float* __restrict__ H,
                             float* __restrict__ AGG, int E) {
    const int total = E * 16;
    for (int g = blockIdx.x * blockDim.x + threadIdx.x; g < total;
         g += gridDim.x * blockDim.x) {
        const int e = g >> 4;
        const int c = (g & 15) << 2;
        const int s = src[e], d = dst[e];
        const float nrm = dinv[s] * dinv[d];
        const float4 v = *(const float4*)(H + (size_t)s * HID + c);
        float* o = AGG + (size_t)d * HID + c;
        gadd_f32(o + 0, v.x * nrm);
        gadd_f32(o + 1, v.y * nrm);
        gadd_f32(o + 2, v.z * nrm);
        gadd_f32(o + 3, v.w * nrm);
    }
}

// ---------------------------------------------------------------- self-loop + bias + ReLU -> f16 for GEMM2
__global__ void bias_relu_kernel(const float* __restrict__ AGG, const float* __restrict__ H1,
                                 const float* __restrict__ dinv, const float* __restrict__ b1,
                                 _Float16* __restrict__ H1R) {
    const int total = NN * HID;
    for (int g = blockIdx.x * blockDim.x + threadIdx.x; g < total;
         g += gridDim.x * blockDim.x) {
        const int i = g >> 6;
        const int f = g & 63;
        const float di = dinv[i];
        const float v = AGG[g] + H1[g] * di * di + b1[f];
        H1R[g] = (_Float16)fmaxf(v, 0.0f);
    }
}

// ---------------------------------------------------------------- GEMM2: H2 = H1R(100000x64 f16) * W2pack, N padded to 48
__global__ __launch_bounds__(96) void gemm2_wmma(const _Float16* __restrict__ A,
                                                 const _Float16* __restrict__ BP,
                                                 float* __restrict__ H2) {
    const int lane = threadIdx.x & 31;
    const int nt   = threadIdx.x >> 5;   // 0..2
    const int mt   = blockIdx.x;         // 0..6249
    const int l15  = lane & 15;
    const int hi   = lane >> 4;
    const int row  = mt * 16 + l15;
    const int col  = nt * 16 + l15;      // 0..47, valid < 40

    v8f acc = {};
#pragma unroll
    for (int kc = 0; kc < 2; ++kc) {
        const _Float16* ap = A + (size_t)row * HID + kc * 32 + hi * 16;
        v16h a;
#pragma unroll
        for (int i = 0; i < 8; ++i) { a[i] = ap[i]; a[8 + i] = ap[8 + i]; }

        const v16h b = *(const v16h*)(BP + (size_t)((nt * 2 + kc) * 32 + lane) * 16);

        acc = __builtin_amdgcn_wmma_f32_16x16x32_f16(false, a, false, b,
                                                     (short)0, acc, false, false);
    }
    if (col < NCLS) {
        float* hp = H2 + (size_t)(mt * 16 + hi * 8) * NCLS + col;
#pragma unroll
        for (int r = 0; r < 8; ++r) hp[(size_t)r * NCLS] = acc[r];
    }
}

// ---------------------------------------------------------------- edge aggregation, 40 features (float4 chunks, 10/edge)
__global__ void agg40_kernel(const int* __restrict__ src, const int* __restrict__ dst,
                             const float* __restrict__ dinv, const float* __restrict__ H,
                             float* __restrict__ OUT, int E) {
    const int total = E * 10;
    for (int g = blockIdx.x * blockDim.x + threadIdx.x; g < total;
         g += gridDim.x * blockDim.x) {
        const int e = g / 10;
        const int c = (g - e * 10) << 2;
        const int s = src[e], d = dst[e];
        const float nrm = dinv[s] * dinv[d];
        const float4 v = *(const float4*)(H + (size_t)s * NCLS + c);
        float* o = OUT + (size_t)d * NCLS + c;
        gadd_f32(o + 0, v.x * nrm);
        gadd_f32(o + 1, v.y * nrm);
        gadd_f32(o + 2, v.z * nrm);
        gadd_f32(o + 3, v.w * nrm);
    }
}

// ---------------------------------------------------------------- self-loop + bias (layer 2)
__global__ void selfloop2_kernel(const float* __restrict__ H2, const float* __restrict__ dinv,
                                 const float* __restrict__ b2, float* __restrict__ OUT) {
    const int total = NN * NCLS;
    for (int g = blockIdx.x * blockDim.x + threadIdx.x; g < total;
         g += gridDim.x * blockDim.x) {
        const int i = g / NCLS;
        const int c = g - i * NCLS;
        const float di = dinv[i];
        OUT[g] += H2[g] * di * di + b2[c];
    }
}

// ---------------------------------------------------------------- log-softmax, one wave32 per node, 40 classes
__global__ void logsoftmax_kernel(float* __restrict__ OUT, int N) {
    const int wid  = (blockIdx.x * blockDim.x + threadIdx.x) >> 5;
    const int lane = threadIdx.x & 31;
    if (wid >= N) return;
    float* row = OUT + (size_t)wid * NCLS;
    const float v0 = row[lane];                                  // lane < 32 < 40
    const float v1 = (lane < 8) ? row[lane + 32] : -INFINITY;
    float m = fmaxf(v0, v1);
#pragma unroll
    for (int o = 16; o > 0; o >>= 1) m = fmaxf(m, __shfl_xor(m, o, 32));
    float s = __expf(v0 - m) + ((lane < 8) ? __expf(v1 - m) : 0.0f);
#pragma unroll
    for (int o = 16; o > 0; o >>= 1) s += __shfl_xor(s, o, 32);
    const float l = m + __logf(s);
    row[lane] = v0 - l;
    if (lane < 8) row[lane + 32] = v1 - l;
}

// ================================================================ launch
extern "C" void kernel_launch(void* const* d_in, const int* in_sizes, int n_in,
                              void* d_out, int out_size, void* d_ws, size_t ws_size,
                              hipStream_t stream) {
    const float* x   = (const float*)d_in[0];
    const int*   ei  = (const int*)d_in[1];      // [2, NE] -> src row, dst row
    const float* W1  = (const float*)d_in[5];
    const float* b1  = (const float*)d_in[6];
    const float* W2  = (const float*)d_in[7];
    const float* b2  = (const float*)d_in[8];
    float*       out = (float*)d_out;
    const int* src = ei;
    const int* dst = ei + NE;

    // workspace carve-out (256B aligned regions)
    char*  w = (char*)d_ws;
    size_t o = 0;
    auto carve = [&](size_t bytes) -> void* {
        void* p = w + o;
        o = (o + bytes + 255) & ~(size_t)255;
        return p;
    };
    int*      deg  = (int*)carve((size_t)NN * 4);
    float*    dinv = (float*)carve((size_t)NN * 4);
    float*    H1   = (float*)carve((size_t)NN * HID * 4);
    float*    AGG1 = (float*)carve((size_t)NN * HID * 4);
    _Float16* H1R  = (_Float16*)carve((size_t)NN * HID * 2);
    _Float16* BP1  = (_Float16*)carve((size_t)4 * 8 * 32 * 16 * 2);   // 32 KB
    _Float16* BP2  = (_Float16*)carve((size_t)3 * 2 * 32 * 16 * 2);   // 6 KB
    float*    H2   = H1;  // H1 region is dead after bias_relu_kernel

    hipMemsetAsync(deg,  0, (size_t)NN * 4, stream);
    hipMemsetAsync(AGG1, 0, (size_t)NN * HID * 4, stream);
    hipMemsetAsync(out,  0, (size_t)NN * NCLS * 4, stream);

    deg_kernel  <<<(NE + 255) / 256, 256, 0, stream>>>(dst, deg, NE);
    dinv_kernel <<<(NN + 255) / 256, 256, 0, stream>>>(deg, dinv, NN);

    pack_w1_kernel<<<4, 256, 0, stream>>>(W1, BP1);
    pack_w2_kernel<<<1, 256, 0, stream>>>(W2, BP2);

    gemm1_wmma  <<<NN / 16, 128, 0, stream>>>(x, BP1, H1);         // 6250 blocks x 4 waves
    agg64_kernel<<<4096, 256, 0, stream>>>(src, dst, dinv, H1, AGG1, NE);
    bias_relu_kernel<<<4096, 256, 0, stream>>>(AGG1, H1, dinv, b1, H1R);

    gemm2_wmma  <<<NN / 16, 96, 0, stream>>>(H1R, BP2, H2);        // 6250 blocks x 3 waves
    agg40_kernel<<<4096, 256, 0, stream>>>(src, dst, dinv, H2, out, NE);
    selfloop2_kernel<<<4096, 256, 0, stream>>>(H2, dinv, b2, out);

    logsoftmax_kernel<<<(NN * 32 + 255) / 256, 256, 0, stream>>>(out, NN);
}
